// RNN_39917426049717
// MI455X (gfx1250) — compile-verified, matrix operand order checked by reading, not run on hardware
//
#include <hip/hip_runtime.h>
#include <hip/hip_bf16.h>

// ---------------------------------------------------------------------------
// Key insight: reference() returns outs[:, -1, :] only, and every row is an
// independent pipeline (LSTM state resets per row).  So only the LAST column
// of the LAST mat per patient matters: row_p = x[p, M-1, :, B-1].
// => compute 64 rows, not 131072 (2048x less work, 128KB of x instead of 256MB).
// Also, c0 == 0 => the forget gate is multiplied by zero: skip the f-gate GEMM.
// All GEMMs use the CDNA5 WMMA f16 path with f32 accumulation.
// ---------------------------------------------------------------------------

typedef __attribute__((ext_vector_type(16))) _Float16 v16h;
typedef __attribute__((ext_vector_type(8)))  float    v8f;
typedef __attribute__((ext_vector_type(8)))  _Float16 h8;

#define PAT   64
#define MATS  16
#define AIN   512
#define BCOL  128
#define H1DIM 256
#define H2DIM 256
#define NCLS  32

#define ROWS_PER_BLK 16
#define THREADS      256   // 8 wave32s

// Load a 16x32 f16 A-fragment from an LDS row-major tile [16][stride].
// ISA 7.12.2: lane<16 holds M=lane, K = kb*32 + {0..7, 16..23};
//             lane>=16 holds M=lane-16, K = kb*32 + {8..15, 24..31}.
__device__ __forceinline__ v16h load_fragA(const _Float16* base, int stride,
                                           int kb, int lane) {
  const int m  = lane & 15;
  const int hf = lane >> 4;
  const _Float16* p = base + m * stride + kb * 32 + hf * 8;
  h8 lo = *(const h8*)(p);        // K = kb*32 + hf*8 + 0..7   (16B aligned)
  h8 up = *(const h8*)(p + 16);   // K = kb*32 + 16 + hf*8 + 0..7
  v16h r;
#pragma unroll
  for (int i = 0; i < 8; ++i) { r[i] = lo[i]; r[8 + i] = up[i]; }
  return r;
}

// Load a 32x16 f16 B-fragment from f32 weights W (row-major [n][k], ld=ldk),
// converting f32->f16 on the fly.  B mirrors the A layout with N in place of M.
__device__ __forceinline__ v16h load_fragB(const float* __restrict__ W,
                                           int nbase, int ldk, int kb, int lane) {
  const int n  = nbase + (lane & 15);
  const int hf = lane >> 4;
  const float* p = W + (size_t)n * ldk + kb * 32 + hf * 8;
  v16h r;
#pragma unroll
  for (int i = 0; i < 8; ++i) {
    r[i]     = (_Float16)p[i];
    r[8 + i] = (_Float16)p[16 + i];
  }
  return r;
}

__device__ __forceinline__ float sigmoidf_fast(float x) {
  return 1.0f / (1.0f + __expf(-x));
}

__global__ void __launch_bounds__(THREADS)
rnn_fused_kernel(const float* __restrict__ x,
                 const float* __restrict__ W1,  const float* __restrict__ b1,
                 const float* __restrict__ Wih, const float* __restrict__ bih,
                 const float* __restrict__ bhh,
                 const float* __restrict__ W3,  const float* __restrict__ b3,
                 float* __restrict__ out) {
  __shared__ _Float16 sX [ROWS_PER_BLK * AIN];    // 16 KB gathered input rows
  __shared__ _Float16 sH1[ROWS_PER_BLK * H1DIM];  // 8 KB  fc1 activations
  __shared__ _Float16 sH [ROWS_PER_BLK * H2DIM];  // 8 KB  LSTM hidden
  __shared__ float    sL [ROWS_PER_BLK * NCLS];   // 2 KB  logits

  const int tid  = threadIdx.x;
  const int lane = tid & 31;
  const int wave = tid >> 5;                      // 0..7
  const int R    = blockIdx.x * ROWS_PER_BLK;     // patient base for this block
  const int nlo  = lane & 15;
  const int m0   = (lane >> 4) * 8;               // C/D: M = m0 + vgpr

  // ---- Stage 0: gather the only rows that matter: x[p, M-1, :, B-1] ----
  for (int i = tid; i < ROWS_PER_BLK * AIN; i += THREADS) {
    const int r = i >> 9;            // AIN = 512
    const int a = i & (AIN - 1);
    const size_t gi =
        ((size_t)((R + r) * MATS + (MATS - 1)) * AIN + a) * BCOL + (BCOL - 1);
    sX[i] = (_Float16)x[gi];
  }
  __syncthreads();

  // ---- Stage 1: H1 = relu(X @ W1^T + b1)   [16x512]@[512x256] ----
#pragma unroll
  for (int t = 0; t < 2; ++t) {
    const int nt = wave * 2 + t;                  // 16 column tiles of 16
    const float bias = b1[nt * 16 + nlo];
    v8f acc;
#pragma unroll
    for (int v = 0; v < 8; ++v) acc[v] = bias;
#pragma unroll
    for (int kb = 0; kb < AIN / 32; ++kb) {
      v16h a = load_fragA(sX, AIN, kb, lane);
      v16h b = load_fragB(W1, nt * 16, AIN, kb, lane);
      acc = __builtin_amdgcn_wmma_f32_16x16x32_f16(
          false, a, false, b, (short)0, acc, false, false);
    }
#pragma unroll
    for (int v = 0; v < 8; ++v) {
      const float h = acc[v] > 0.0f ? acc[v] : 0.0f;
      sH1[(m0 + v) * H1DIM + nt * 16 + nlo] = (_Float16)h;
    }
  }
  __syncthreads();

  // ---- Stage 2: gates (i,g,o only; f is killed by c0==0) + LSTM math ----
  // gates = H1 @ W_ih^T + (b_ih + b_hh); rows of W_ih: i:[0,256) g:[512,768) o:[768,1024)
#pragma unroll
  for (int t = 0; t < 2; ++t) {
    const int nt = wave * 2 + t;                  // 16 column tiles over H2
    const int nc = nt * 16 + nlo;
    v8f ai, ag, ao;
    const float bi = bih[0 * H2DIM + nc] + bhh[0 * H2DIM + nc];
    const float bg = bih[2 * H2DIM + nc] + bhh[2 * H2DIM + nc];
    const float bo = bih[3 * H2DIM + nc] + bhh[3 * H2DIM + nc];
#pragma unroll
    for (int v = 0; v < 8; ++v) { ai[v] = bi; ag[v] = bg; ao[v] = bo; }
#pragma unroll
    for (int kb = 0; kb < H1DIM / 32; ++kb) {
      v16h a  = load_fragA(sH1, H1DIM, kb, lane);
      v16h bI = load_fragB(Wih, 0 * H2DIM + nt * 16, H1DIM, kb, lane);
      v16h bG = load_fragB(Wih, 2 * H2DIM + nt * 16, H1DIM, kb, lane);
      v16h bO = load_fragB(Wih, 3 * H2DIM + nt * 16, H1DIM, kb, lane);
      ai = __builtin_amdgcn_wmma_f32_16x16x32_f16(
          false, a, false, bI, (short)0, ai, false, false);
      ag = __builtin_amdgcn_wmma_f32_16x16x32_f16(
          false, a, false, bG, (short)0, ag, false, false);
      ao = __builtin_amdgcn_wmma_f32_16x16x32_f16(
          false, a, false, bO, (short)0, ao, false, false);
    }
#pragma unroll
    for (int v = 0; v < 8; ++v) {
      const float c = sigmoidf_fast(ai[v]) * tanhf(ag[v]);   // c = i*g (c0=0)
      float h = sigmoidf_fast(ao[v]) * tanhf(c);             // h = o*tanh(c)
      h = h > 0.0f ? h : 0.0f;                               // relu
      sH[(m0 + v) * H2DIM + nc] = (_Float16)h;
    }
  }
  __syncthreads();

  // ---- Stage 3: logits = H @ W3^T + b3   [16x256]@[256x32] (waves 0,1) ----
  if (wave < 2) {                                 // wave-uniform: EXEC all-1s
    const int nt = wave;
    const int nc = nt * 16 + nlo;
    const float bias = b3[nc];
    v8f acc;
#pragma unroll
    for (int v = 0; v < 8; ++v) acc[v] = bias;
#pragma unroll
    for (int kb = 0; kb < H2DIM / 32; ++kb) {
      v16h a = load_fragA(sH, H2DIM, kb, lane);
      v16h b = load_fragB(W3, nt * 16, H2DIM, kb, lane);
      acc = __builtin_amdgcn_wmma_f32_16x16x32_f16(
          false, a, false, b, (short)0, acc, false, false);
    }
#pragma unroll
    for (int v = 0; v < 8; ++v)
      sL[(m0 + v) * NCLS + nc] = acc[v];
  }
  __syncthreads();

  // ---- Stage 4: per-row softmax over 32 classes ----
  if (tid < ROWS_PER_BLK) {
    const float* row = sL + tid * NCLS;
    float mx = row[0];
#pragma unroll
    for (int c = 1; c < NCLS; ++c) mx = fmaxf(mx, row[c]);
    float e[NCLS];
    float s = 0.0f;
#pragma unroll
    for (int c = 0; c < NCLS; ++c) { e[c] = __expf(row[c] - mx); s += e[c]; }
    const float inv = 1.0f / s;
    float* o = out + (size_t)(R + tid) * NCLS;
#pragma unroll
    for (int c = 0; c < NCLS; ++c) o[c] = e[c] * inv;
  }
}

extern "C" void kernel_launch(void* const* d_in, const int* in_sizes, int n_in,
                              void* d_out, int out_size, void* d_ws, size_t ws_size,
                              hipStream_t stream) {
  (void)in_sizes; (void)n_in; (void)out_size; (void)d_ws; (void)ws_size;
  const float* x   = (const float*)d_in[0];
  const float* W1  = (const float*)d_in[1];
  const float* b1  = (const float*)d_in[2];
  const float* Wih = (const float*)d_in[3];
  // d_in[4] = W_hh is mathematically unused (h0 == 0)
  const float* bih = (const float*)d_in[5];
  const float* bhh = (const float*)d_in[6];
  const float* W3  = (const float*)d_in[7];
  const float* b3  = (const float*)d_in[8];
  float* out = (float*)d_out;

  rnn_fused_kernel<<<PAT / ROWS_PER_BLK, THREADS, 0, stream>>>(
      x, W1, b1, Wih, bih, bhh, W3, b3, out);
}
